// Attention_5669356833508
// MI455X (gfx1250) — matrix operand order repaired and checked by reading, not compiled, not run on hardware
//
#include <hip/hip_runtime.h>
#include <hip/hip_bf16.h>

typedef __bf16 v16bf __attribute__((ext_vector_type(16)));
typedef __bf16 v8bf  __attribute__((ext_vector_type(8)));
typedef float  v8f   __attribute__((ext_vector_type(8)));
typedef unsigned int u32x4 __attribute__((ext_vector_type(4)));
typedef int          i32x4 __attribute__((ext_vector_type(4)));
typedef int          i32x8 __attribute__((ext_vector_type(8)));

#define HW 16384
#define IMW 128

// fp32 -> bf16, round-to-nearest-even, pure integer ops
__device__ __forceinline__ __bf16 f2bf(float f) {
    unsigned u = __builtin_bit_cast(unsigned, f);
    unsigned r = (u + 0x7FFFu + ((u >> 16) & 1u)) >> 16;
    unsigned short s = (unsigned short)r;
    return __builtin_bit_cast(__bf16, s);
}
__device__ __forceinline__ float bf2f(__bf16 b) {
    unsigned short s = __builtin_bit_cast(unsigned short, b);
    unsigned u = (unsigned)s << 16;
    return __builtin_bit_cast(float, u);
}

// Generic pointer to shared: low 32 bits are the LDS byte offset (shared aperture
// occupies the high bits), which is what TDM lds_addr / async VDST expect.
__device__ __forceinline__ unsigned lds_off(const void* p) {
    return (unsigned)(uintptr_t)p;
}

// ---------------------------------------------------------------------------
// Tensor Data Mover: 2D bf16 tile load, global (strided rows) -> LDS (packed
// rows, tile_dim0 elements each). D# per CDNA5 ISA.
// 5-arg builtin on ROCm 7.2 (clang-22), 6-arg on clang-23 toolchain.
// ---------------------------------------------------------------------------
__device__ __forceinline__ void tdm_load_2d_bf16(unsigned lds_byte_addr,
                                                 const void* gaddr,
                                                 unsigned tile_rows,
                                                 unsigned tile_cols,
                                                 unsigned row_stride_elems,
                                                 unsigned tensor_rows) {
    unsigned long long ga = (unsigned long long)(uintptr_t)gaddr;
    u32x4 g0;
    g0.x = 1u;                                    // count=1, user descriptor
    g0.y = lds_byte_addr;                         // lds_addr  [63:32]
    g0.z = (unsigned)(ga & 0xFFFFFFFFu);          // global_addr[31:0]  -> bits 95:64
    g0.w = (unsigned)((ga >> 32) & 0x1FFFFFFu)    // global_addr[56:32] -> bits 120:96
         | (2u << 30);                            // type = 2 ("image")
    i32x8 g1;
    g1[0] = (int)(1u << 16);                      // data_size=1 (2 bytes/elem)
    g1[1] = (int)((row_stride_elems & 0xFFFFu) << 16);        // tensor_dim0[15:0]
    g1[2] = (int)(((row_stride_elems >> 16) & 0xFFFFu)        // tensor_dim0[31:16]
                  | ((tensor_rows & 0xFFFFu) << 16));         // tensor_dim1[15:0]
    g1[3] = (int)(((tensor_rows >> 16) & 0xFFFFu)
                  | ((tile_cols & 0xFFFFu) << 16));           // tile_dim0
    g1[4] = (int)(tile_rows & 0xFFFFu);                       // tile_dim1 (dim2=0)
    g1[5] = (int)row_stride_elems;                            // tensor_dim0_stride
    g1[6] = 0;
    g1[7] = 0;
    i32x4 z4 = {0, 0, 0, 0};
#if defined(__clang_major__) && __clang_major__ >= 23
    i32x8 z8 = {0, 0, 0, 0, 0, 0, 0, 0};
    __builtin_amdgcn_tensor_load_to_lds(g0, g1, z4, z4, z8, 0);
#else
    __builtin_amdgcn_tensor_load_to_lds(g0, g1, z4, z4, 0);
#endif
}

// A-fragment (16x32 bf16) from LDS tile stored [row][k].
__device__ __forceinline__ v16bf load_frag_a(const __bf16* base, int row0, int k0,
                                             int lane, int stride) {
    int row = row0 + (lane & 15);
    int kb  = k0 + ((lane >> 4) << 3);
    const __bf16* p = base + row * stride + kb;
    v8bf lo = *(const v8bf*)p;
    v8bf hi = *(const v8bf*)(p + 16);
    return __builtin_shufflevector(lo, hi, 0,1,2,3,4,5,6,7,8,9,10,11,12,13,14,15);
}

// B-fragment (32x16 bf16) from LDS tile stored [n][k].
__device__ __forceinline__ v16bf load_frag_b(const __bf16* base, int n0, int k0,
                                             int lane, int stride) {
    int n  = n0 + (lane & 15);
    int kb = k0 + ((lane >> 4) << 3);
    const __bf16* p = base + n * stride + kb;
    v8bf lo = *(const v8bf*)p;
    v8bf hi = *(const v8bf*)(p + 16);
    return __builtin_shufflevector(lo, hi, 0,1,2,3,4,5,6,7,8,9,10,11,12,13,14,15);
}

__device__ __forceinline__ v8f wmma_bf16(v16bf a, v16bf b, v8f c) {
    return __builtin_amdgcn_wmma_f32_16x16x32_bf16(false, a, false, b, (short)0, c,
                                                   false, false);
}

// ---------------------------------------------------------------------------
// 1x1 conv as GEMM: Out[b][m][pix] = sum_k W[m][k] * X[b][k][pix], K = 192.
// ---------------------------------------------------------------------------
__global__ void __launch_bounds__(256)
pw_gemm_wmma(const float* __restrict__ Wt, const float* __restrict__ Xin,
             float* __restrict__ Out, int M) {
    const int K = 192;
    int mtile = blockIdx.x * 64;
    long ncol = (long)blockIdx.y * 64;
    int b   = (int)(ncol / HW);
    int pix = (int)(ncol % HW);
    const float* Xb = Xin + (size_t)b * K * HW;

    __shared__ __align__(16) __bf16 As[64][192];   // [m][k]
    __shared__ __align__(16) __bf16 Bs[64][192];   // [n][k]

    int tid = threadIdx.x, lane = tid & 31, wid = tid >> 5;
    for (int idx = tid; idx < 64 * K; idx += 256) {
        int r = idx / K, c = idx - r * K;
        As[r][c] = f2bf(Wt[(size_t)(mtile + r) * K + c]);
    }
    for (int idx = tid; idx < K * 64; idx += 256) {
        int k = idx >> 6, n = idx & 63;
        Bs[n][k] = f2bf(Xb[(size_t)k * HW + pix + n]);
    }
    __syncthreads();

    int wm = (wid & 3) * 16;
    int wn = (wid >> 2) * 32;
    v8f acc0 = {}, acc1 = {};
    for (int k0 = 0; k0 < K; k0 += 32) {
        v16bf a = load_frag_a(&As[0][0], wm, k0, lane, 192);
        acc0 = wmma_bf16(a, load_frag_b(&Bs[0][0], wn,      k0, lane, 192), acc0);
        acc1 = wmma_bf16(a, load_frag_b(&Bs[0][0], wn + 16, k0, lane, 192), acc1);
    }
    int n = lane & 15;
    for (int v = 0; v < 8; ++v) {
        int row = mtile + wm + v + ((lane >> 4) << 3);
        size_t base = ((size_t)b * M + row) * (size_t)HW + pix;
        Out[base + wn + n]      = acc0[v];
        Out[base + wn + 16 + n] = acc1[v];
    }
}

// ---------------------------------------------------------------------------
// Depthwise 3x3, pad 1. Writes bf16 so downstream stages can DMA raw tiles.
// ---------------------------------------------------------------------------
__global__ void __launch_bounds__(256)
dwconv3x3(const float* __restrict__ in, const float* __restrict__ w,
          __bf16* __restrict__ out) {
    size_t idx = (size_t)blockIdx.x * 256 + threadIdx.x;
    int pix = (int)(idx & (HW - 1));
    int ch  = (int)((idx >> 14) % 576);
    int y = pix >> 7, x = pix & (IMW - 1);
    const float* p  = in + (idx - pix);
    const float* wc = w + ch * 9;
    float s = 0.f;
    #pragma unroll
    for (int dy = -1; dy <= 1; ++dy) {
        int yy = y + dy;
        if (yy < 0 || yy >= IMW) continue;
        #pragma unroll
        for (int dx = -1; dx <= 1; ++dx) {
            int xx = x + dx;
            if (xx < 0 || xx >= IMW) continue;
            s += p[yy * IMW + xx] * wc[(dy + 1) * 3 + (dx + 1)];
        }
    }
    out[idx] = f2bf(s);
}

// ---------------------------------------------------------------------------
// Row L2 norms for q,k (channels 0..383 per batch), bf16 input.
// ---------------------------------------------------------------------------
__global__ void __launch_bounds__(256)
row_norms(const __bf16* __restrict__ qkv2, float* __restrict__ norms) {
    int r = blockIdx.x;
    int b = r / 384, j = r % 384;
    const __bf16* p = qkv2 + ((size_t)b * 576 + j) * HW;
    float s = 0.f;
    for (int i = threadIdx.x; i < HW; i += 256) { float v = bf2f(p[i]); s += v * v; }
    __shared__ float red[256];
    red[threadIdx.x] = s; __syncthreads();
    for (int off = 128; off; off >>= 1) {
        if ((int)threadIdx.x < off) red[threadIdx.x] += red[threadIdx.x + off];
        __syncthreads();
    }
    if (threadIdx.x == 0) norms[r] = sqrtf(red[0]);
}

// ---------------------------------------------------------------------------
// attn = softmax((q/||q||)·(k/||k||)^T * temp). Q/K tiles arrive via TDM.
// 4 waves, 128 threads; M=48 (pad 64), N=48, K=16384.
// ---------------------------------------------------------------------------
__global__ void __launch_bounds__(128)
qk_softmax(const __bf16* __restrict__ qkv2, const float* __restrict__ norms,
           const float* __restrict__ temp, __bf16* __restrict__ attnOut) {
    int bh = blockIdx.x, b = bh >> 2, h = bh & 3;
    const __bf16* qp = qkv2 + ((size_t)b * 576 + h * 48) * HW;
    const __bf16* kp = qkv2 + ((size_t)b * 576 + 192 + h * 48) * HW;

    __shared__ __align__(16) __bf16 Qs[64][128];   // rows 48..63 zero pad
    __shared__ __align__(16) __bf16 Ks[48][128];
    __shared__ float attnS[48][48];

    int tid = threadIdx.x, lane = tid & 31, w = tid >> 5;
    // zero the pad rows once (disjoint from TDM target rows 0..47)
    for (int idx = tid; idx < 16 * 128; idx += 128)
        Qs[48 + (idx >> 7)][idx & 127] = f2bf(0.f);

    unsigned ldsQ = lds_off(&Qs[0][0]);
    unsigned ldsK = lds_off(&Ks[0][0]);

    v8f acc[3] = {};
    for (int nc = 0; nc < HW; nc += 128) {
        if (nc + 128 < HW) {
            __builtin_prefetch(qp + nc + 128, 0, 0);   // global_prefetch_b8
            __builtin_prefetch(kp + nc + 128, 0, 0);
        }
        if (w == 0) {   // one wave drives the DMA engine
            tdm_load_2d_bf16(ldsQ, qp + nc, 48, 128, HW, 48);
            tdm_load_2d_bf16(ldsK, kp + nc, 48, 128, HW, 48);
            __builtin_amdgcn_s_wait_tensorcnt(0);
        }
        __syncthreads();
        for (int k0 = 0; k0 < 128; k0 += 32) {
            v16bf a = load_frag_a(&Qs[0][0], w * 16, k0, lane, 128);
            #pragma unroll
            for (int j = 0; j < 3; ++j)
                acc[j] = wmma_bf16(a, load_frag_b(&Ks[0][0], j * 16, k0, lane, 128),
                                   acc[j]);
        }
        __syncthreads();
    }
    float t = temp[h];
    const float* nq = norms + b * 384 + h * 48;
    const float* nk = norms + b * 384 + 192 + h * 48;
    for (int j = 0; j < 3; ++j) {
        int col = j * 16 + (lane & 15);
        for (int v = 0; v < 8; ++v) {
            int row = w * 16 + v + ((lane >> 4) << 3);
            if (row < 48) {
                float dq = fmaxf(nq[row], 1e-12f);
                float dk = fmaxf(nk[col], 1e-12f);
                attnS[row][col] = acc[j][v] / (dq * dk) * t;
            }
        }
    }
    __syncthreads();
    if (tid < 48) {
        float m = -1e30f;
        for (int c = 0; c < 48; ++c) m = fmaxf(m, attnS[tid][c]);
        float s = 0.f;
        for (int c = 0; c < 48; ++c) s += __expf(attnS[tid][c] - m);
        float inv = 1.f / s;
        for (int c = 0; c < 48; ++c)
            attnOut[(size_t)bh * 2304 + tid * 48 + c] =
                f2bf(__expf(attnS[tid][c] - m) * inv);
    }
}

// ---------------------------------------------------------------------------
// out = attn · v. attn (48x48 bf16) arrives via global_load_async_to_lds_b128.
// ---------------------------------------------------------------------------
__global__ void __launch_bounds__(256)
attn_v(const __bf16* __restrict__ qkv2, const __bf16* __restrict__ attnBuf,
       float* __restrict__ aout) {
    int bh = blockIdx.x, b = bh >> 2, h = bh & 3;
    int nc0 = blockIdx.y * 256;
    const __bf16* vp = qkv2 + ((size_t)b * 576 + 384 + h * 48) * HW;

    __shared__ __align__(16) __bf16 Aat[64][64];   // attn [m][d], zero-padded
    __shared__ __align__(16) __bf16 Bv[256][64];   // v^T [n][d], zero-padded

    int tid = threadIdx.x, lane = tid & 31, wid = tid >> 5;
    // zero only the padding cells (disjoint from the async-DMA target region)
    for (int idx = tid; idx < 64 * 64; idx += 256) {
        int r = idx >> 6, c = idx & 63;
        if (r >= 48 || c >= 48) Aat[r][c] = f2bf(0.f);
    }
    // async copy 48x48 bf16 attn (4608 B) into strided LDS rows, 16 B per lane
    const char* gsrc = (const char*)(attnBuf + (size_t)bh * 2304);
    for (int f = tid * 16; f < 4608; f += 4096) {
        int r = f / 96, c = f - r * 96;               // 96 B per source row
        unsigned lds = lds_off(&Aat[r][0]) + (unsigned)c;
        asm volatile("global_load_async_to_lds_b128 %0, %1, %2"
                     :: "v"(lds), "v"(f), "s"(gsrc) : "memory");
    }
    for (int idx = tid; idx < 64 * 256; idx += 256) {
        int d = idx >> 8, n = idx & 255;
        Bv[n][d] = (d < 48) ? vp[(size_t)d * HW + nc0 + n] : f2bf(0.f);
    }
    asm volatile("s_wait_asynccnt 0x0" ::: "memory");
    __syncthreads();

    int wm  = (wid & 3) * 16;
    int wnb = (wid >> 2) * 128;
    v16bf a0 = load_frag_a(&Aat[0][0], wm, 0,  lane, 64);
    v16bf a1 = load_frag_a(&Aat[0][0], wm, 32, lane, 64);
    for (int j = 0; j < 8; ++j) {
        int n0 = wnb + j * 16;
        v8f acc = {};
        acc = wmma_bf16(a0, load_frag_b(&Bv[0][0], n0, 0,  lane, 64), acc);
        acc = wmma_bf16(a1, load_frag_b(&Bv[0][0], n0, 32, lane, 64), acc);
        int col = nc0 + n0 + (lane & 15);
        for (int v = 0; v < 8; ++v) {
            int row = wm + v + ((lane >> 4) << 3);
            if (row < 48)
                aout[((size_t)(b * 192) + h * 48 + row) * HW + col] = acc[v];
        }
    }
}

// ---------------------------------------------------------------------------
extern "C" void kernel_launch(void* const* d_in, const int* in_sizes, int n_in,
                              void* d_out, int out_size, void* d_ws, size_t ws_size,
                              hipStream_t stream) {
    const float* x      = (const float*)d_in[0];   // [8,192,128,128]
    const float* qkv_w  = (const float*)d_in[1];   // [576,192,1,1]
    const float* dw_w   = (const float*)d_in[2];   // [576,1,3,3]
    const float* proj_w = (const float*)d_in[3];   // [192,192,1,1]
    const float* temp   = (const float*)d_in[4];   // [4,1,1]
    float* out = (float*)d_out;

    const size_t szQkv  = (size_t)8 * 576 * HW * sizeof(float);    // f32 qkv
    const size_t szQkvB = (size_t)8 * 576 * HW * sizeof(__bf16);   // bf16 qkv2
    char* wsb = (char*)d_ws;
    float*  qkv   = (float*)(wsb);
    __bf16* qkv2b = (__bf16*)(wsb + szQkv);
    float*  norms = (float*)(wsb + szQkv + szQkvB);
    __bf16* attnb = (__bf16*)(wsb + szQkv + szQkvB + 8 * 384 * sizeof(float));
    float*  aout  = (float*)(wsb + szQkv + szQkvB + 8 * 384 * sizeof(float)
                                  + (size_t)32 * 2304 * sizeof(__bf16));

    pw_gemm_wmma<<<dim3(9, 2048), 256, 0, stream>>>(qkv_w, x, qkv, 576);
    dwconv3x3<<<(8 * 576 * HW) / 256, 256, 0, stream>>>(qkv, dw_w, qkv2b);
    row_norms<<<8 * 384, 256, 0, stream>>>(qkv2b, norms);
    qk_softmax<<<32, 128, 0, stream>>>(qkv2b, norms, temp, attnb);
    attn_v<<<dim3(32, 64), 256, 0, stream>>>(qkv2b, attnb, aout);
    pw_gemm_wmma<<<dim3(3, 2048), 256, 0, stream>>>(proj_w, aout, out, 192);
}
